// EMAVectorQuantizer_50053548867723
// MI455X (gfx1250) — compile-verified, matrix operand order checked by reading, not run on hardware
//
#include <hip/hip_runtime.h>

// ---------------------------------------------------------------------------
// EMA Vector Quantizer for MI455X (gfx1250, wave32, WMMA + TDM)
//
// Dominant cost: 32768x8192x64 distance argmin (34.4 GFLOP, compute-bound).
// - Codebook pre-rounded to bf16 (prep kernel); norms computed from the
//   rounded rows -> argmin is EXACT over the bf16 codebook.
// - z kept exact via hi/lo bf16 split: z.e = z_hi.e + z_lo.e (2 WMMA terms).
// - 32 rows/wave (2 row-tiles share every B load) -> 8 WMMAs per 4 ds_b128.
// - Embedding super-tiles (128 codes) staged in LDS by the Tensor Data Mover
//   with pad_enable (128B row + 16B pad = 144B pitch, bank-spread), double
//   buffered with s_wait_tensorcnt; fallback to cooperative loads if the
//   TDM builtin is unavailable. This toolchain exposes the 6-arg TDM builtin
//   (amdgpu-toolchain / clang-23): extra zero int32x8 group passed.
// ---------------------------------------------------------------------------

typedef __attribute__((ext_vector_type(16))) __bf16 v16bf;
typedef __attribute__((ext_vector_type(8)))  float  v8f;
typedef __attribute__((ext_vector_type(4)))  unsigned int u32x4;
typedef __attribute__((ext_vector_type(8)))  int          i32x8;
typedef __attribute__((ext_vector_type(4)))  int          i32x4;

#define NUM_CODES 8192
#define CODE_DIM  64
#define N_ROWS    32768
#define DECAY     0.99f
#define EPSILON   1e-5f
#define COMMIT    0.25f

// workspace layout (float offsets)
#define WS_LOSS   0
#define WS_NSUM   1
#define WS_COUNTS 2
#define WS_NCS    (WS_COUNTS + NUM_CODES)
#define WS_DW     (WS_NCS + NUM_CODES)
#define WS_NORM   (WS_DW + NUM_CODES * CODE_DIM)
#define WS_IDX    (WS_NORM + NUM_CODES)          // stored as int
#define WS_ZERON  WS_NORM                        // zero [0, WS_NORM)
#define WS_FLOATS (WS_IDX + N_ROWS)
#define WS_EHI_BYTE (((WS_FLOATS * 4) + 255) & ~255)   // bf16 codebook region

// output layout (float offsets into d_out)
#define O_ZQ   0
#define O_IDX  (O_ZQ + N_ROWS * CODE_DIM)
#define O_LOSS (O_IDX + N_ROWS)
#define O_EMB  (O_LOSS + 1)
#define O_CS   (O_EMB + NUM_CODES * CODE_DIM)
#define O_EEE  (O_CS + NUM_CODES)

// ---------------------------------------------------------------------------
__global__ void vq_zero_kernel(float* __restrict__ p, int n) {
    int i = blockIdx.x * blockDim.x + threadIdx.x;
    if (i < n) p[i] = 0.0f;
}

// round codebook to bf16 + squared norms of the ROUNDED rows (consistent
// metric: argmin of ||z - e_bf16||^2 is exact)
__global__ void vq_prep_kernel(const float* __restrict__ emb,
                               __bf16* __restrict__ ehi,
                               float* __restrict__ norm2) {
    int k = blockIdx.x * blockDim.x + threadIdx.x;
    if (k >= NUM_CODES) return;
    const float* e = emb + (size_t)k * CODE_DIM;
    __bf16* o = ehi + (size_t)k * CODE_DIM;
    float s = 0.0f;
#pragma unroll
    for (int d = 0; d < CODE_DIM; ++d) {
        __bf16 r = (__bf16)e[d];
        o[d] = r;
        float rf = (float)r;
        s = fmaf(rf, rf, s);
    }
    norm2[k] = s;
}

// ---------------------------------------------------------------------------
#define SUP  128         // codes per LDS super-tile
#define EPAD 72          // padded bf16 row pitch (144B: 16B aligned, bank-spread)
#define NSUP (NUM_CODES / SUP)

#if __has_builtin(__builtin_amdgcn_tensor_load_to_lds) && \
    __has_builtin(__builtin_amdgcn_s_wait_tensorcnt)
#define USE_TDM 1
#else
#define USE_TDM 0
#endif

__global__ __launch_bounds__(256) void vq_argmin_kernel(
    const float* __restrict__ z, const __bf16* __restrict__ ehi_g,
    const float* __restrict__ normg, int* __restrict__ idx_out,
    float* __restrict__ idx_f_out) {
    __shared__ __bf16 sh_e[2][SUP * EPAD];
    __shared__ float  sh_norm[2][SUP];

    const int lane = threadIdx.x & 31;
    const int wave = threadIdx.x >> 5;
    const int n16  = lane & 15;   // M (for A) / N (for B) within 16x16 tile
    const int h    = lane >> 4;   // lane-half selects K sub-block
    const int rowbase = (blockIdx.x * 8 + wave) * 32;

    union AB { v16bf v; __bf16 e[16]; };

    // Load this wave's 32 z rows once; split into bf16 hi + residual lo.
    // ISA 16-bit A 16x32 layout: lane = M%16; half h: V0-3 <- K=h*8..+7,
    // V4-7 <- K=16+h*8..+7  (chunk c adds c*32 to K).
    AB a_hi[2][2], a_lo[2][2];   // [row-tile][k-chunk]
#pragma unroll
    for (int rt = 0; rt < 2; ++rt) {
        const float* zrow = z + (size_t)(rowbase + rt * 16 + n16) * CODE_DIM;
#pragma unroll
        for (int c = 0; c < 2; ++c)
#pragma unroll
            for (int p = 0; p < 2; ++p) {
                int k0 = c * 32 + p * 16 + h * 8;
#pragma unroll
                for (int i = 0; i < 8; ++i) {
                    float v = zrow[k0 + i];
                    __bf16 hi = (__bf16)v;
                    a_hi[rt][c].e[p * 8 + i] = hi;
                    a_lo[rt][c].e[p * 8 + i] = (__bf16)(v - (float)hi);
                }
            }
    }

    float dmin[2][8];
    int   amin[2][8];
#pragma unroll
    for (int rt = 0; rt < 2; ++rt)
#pragma unroll
        for (int j = 0; j < 8; ++j) { dmin[rt][j] = 3.4e38f; amin[rt][j] = 0; }

#if USE_TDM
#define USE_TDM_PATH 1
#define TDM_LOAD(g0, g1, g2, g3, g4) \
    __builtin_amdgcn_tensor_load_to_lds(g0, g1, g2, g3, g4, 0)
#define TDM_WAIT() __builtin_amdgcn_s_wait_tensorcnt(0)
#else
#define USE_TDM_PATH 0
#define TDM_LOAD(g0, g1, g2, g3, g4) ((void)0)
#define TDM_WAIT() ((void)0)
#endif

    // ---- staging: TDM descriptor = 2D tile, 128 rows x 64 bf16, row pitch
    // 64 elems in memory, +16B LDS pad per 128B row -> 144B pitch (= EPAD*2).
#define STAGE(bufi, sb)                                                        \
    do {                                                                       \
        if (threadIdx.x < SUP) sh_norm[bufi][threadIdx.x] = normg[(sb) + threadIdx.x]; \
        if (USE_TDM_PATH) {                                                    \
            if (wave == 0) {                                                   \
                unsigned long long ga =                                        \
                    (unsigned long long)(size_t)(ehi_g + (size_t)(sb) * CODE_DIM); \
                unsigned lds_off = (unsigned)(size_t)(void*)&sh_e[bufi][0];    \
                u32x4 g0;                                                      \
                g0.x = 1u;                          /* count=1 */              \
                g0.y = lds_off;                     /* lds_addr */             \
                g0.z = (unsigned)(ga & 0xFFFFFFFFu);                           \
                g0.w = (unsigned)((ga >> 32) & 0x01FFFFFFu) | (2u << 30);      \
                i32x8 g1;                                                      \
                g1[0] = (1 << 16)    /* data_size = 2B */                      \
                      | (1 << 20)    /* pad_enable */                          \
                      | (4 << 22)    /* pad_interval: 32 DWORDs = 128B */      \
                      | (3 << 25);   /* pad_amount: 4 DWORDs = 16B  */         \
                g1[1] = (CODE_DIM & 0xFFFF) << 16;  /* tensor_dim0 = 64 */     \
                g1[2] = (NUM_CODES & 0xFFFF) << 16; /* tensor_dim1 */          \
                g1[3] = (CODE_DIM & 0xFFFF) << 16;  /* tile_dim0 = 64 */       \
                g1[4] = SUP;                        /* tile_dim1 = 128 */      \
                g1[5] = CODE_DIM;                   /* tensor_dim0_stride */   \
                g1[6] = 0;                                                     \
                g1[7] = 0;                                                     \
                i32x4 g2 = {0, 0, 0, 0};                                       \
                i32x4 g3 = {0, 0, 0, 0};                                       \
                i32x8 g4 = {0, 0, 0, 0, 0, 0, 0, 0};                           \
                TDM_LOAD(g0, g1, g2, g3, g4);                                  \
            }                                                                  \
        } else {                                                               \
            for (int i = threadIdx.x; i < SUP * CODE_DIM; i += 256) {          \
                int c = i >> 6, k = i & 63;                                    \
                sh_e[bufi][c * EPAD + k] = ehi_g[(size_t)((sb) + c) * CODE_DIM + k]; \
            }                                                                  \
        }                                                                      \
    } while (0)

    STAGE(0, 0);
    for (int s = 0; s < NSUP; ++s) {
        const int buf = s & 1;
        const int sbase = s * SUP;
        if (USE_TDM_PATH) { if (wave == 0) TDM_WAIT(); }
        __syncthreads();   // staged data visible; prior compute on buf done
        if (s + 1 < NSUP) STAGE(buf ^ 1, sbase + SUP);

        for (int t = 0; t < SUP / 16; ++t) {
            const int cl = t * 16 + n16;           // local code for this lane
            const __bf16* pe = &sh_e[buf][cl * EPAD];
            AB bh[2];
#pragma unroll
            for (int c = 0; c < 2; ++c)
#pragma unroll
                for (int p = 0; p < 2; ++p) {
                    int k0 = c * 32 + p * 16 + h * 8;
#pragma unroll
                    for (int i = 0; i < 8; ++i) bh[c].e[p * 8 + i] = pe[k0 + i];
                }

            v8f acc0 = {}, acc1 = {};
            // z.e = z_hi.e + z_lo.e, chained over K=64; alternate
            // accumulators to avoid back-to-back WMMA RAW hazards
            acc0 = __builtin_amdgcn_wmma_f32_16x16x32_bf16(false, a_hi[0][0].v, false, bh[0].v, (short)0, acc0, false, false);
            acc1 = __builtin_amdgcn_wmma_f32_16x16x32_bf16(false, a_hi[1][0].v, false, bh[0].v, (short)0, acc1, false, false);
            acc0 = __builtin_amdgcn_wmma_f32_16x16x32_bf16(false, a_hi[0][1].v, false, bh[1].v, (short)0, acc0, false, false);
            acc1 = __builtin_amdgcn_wmma_f32_16x16x32_bf16(false, a_hi[1][1].v, false, bh[1].v, (short)0, acc1, false, false);
            acc0 = __builtin_amdgcn_wmma_f32_16x16x32_bf16(false, a_lo[0][0].v, false, bh[0].v, (short)0, acc0, false, false);
            acc1 = __builtin_amdgcn_wmma_f32_16x16x32_bf16(false, a_lo[1][0].v, false, bh[0].v, (short)0, acc1, false, false);
            acc0 = __builtin_amdgcn_wmma_f32_16x16x32_bf16(false, a_lo[0][1].v, false, bh[1].v, (short)0, acc0, false, false);
            acc1 = __builtin_amdgcn_wmma_f32_16x16x32_bf16(false, a_lo[1][1].v, false, bh[1].v, (short)0, acc1, false, false);

            const float nn = sh_norm[buf][cl];
            const int codeIdx = sbase + cl;
            // C/D layout: VGPR j, lanes 0-15 -> M=j, lanes 16-31 -> M=j+8
#pragma unroll
            for (int j = 0; j < 8; ++j) {
                float d0 = fmaf(-2.0f, acc0[j], nn);   // ||z||^2 const: dropped
                bool p0 = d0 < dmin[0][j];
                dmin[0][j] = p0 ? d0 : dmin[0][j];
                amin[0][j] = p0 ? codeIdx : amin[0][j];
                float d1 = fmaf(-2.0f, acc1[j], nn);
                bool p1 = d1 < dmin[1][j];
                dmin[1][j] = p1 ? d1 : dmin[1][j];
                amin[1][j] = p1 ? codeIdx : amin[1][j];
            }
        }
        __syncthreads();   // all waves done with buf before its next DMA
    }

    // reduce across the 16 lanes holding different N for the same rows;
    // tie-break toward the lowest code index (argmin-first semantics)
#pragma unroll
    for (int rt = 0; rt < 2; ++rt) {
#pragma unroll
        for (int j = 0; j < 8; ++j) {
#pragma unroll
            for (int off = 8; off >= 1; off >>= 1) {
                float om = __shfl_xor(dmin[rt][j], off, 32);
                int   oa = __shfl_xor(amin[rt][j], off, 32);
                bool pr = (om < dmin[rt][j]) || (om == dmin[rt][j] && oa < amin[rt][j]);
                dmin[rt][j] = pr ? om : dmin[rt][j];
                amin[rt][j] = pr ? oa : amin[rt][j];
            }
        }
    }
    if (n16 == 0) {
#pragma unroll
        for (int rt = 0; rt < 2; ++rt)
#pragma unroll
            for (int j = 0; j < 8; ++j) {
                int row = rowbase + rt * 16 + h * 8 + j;
                idx_out[row]   = amin[rt][j];
                idx_f_out[row] = (float)amin[rt][j];
            }
    }
}

// ---------------------------------------------------------------------------
// gather z_q, vq-loss partial sums, scatter counts/dw via global f32 atomics
__global__ __launch_bounds__(256) void vq_scatter_kernel(
    const float* __restrict__ z, const float* __restrict__ emb,
    const int* __restrict__ idx, float* __restrict__ out_zq,
    float* __restrict__ counts, float* __restrict__ dw,
    float* __restrict__ loss_accum) {
    __shared__ float red[256];
    int gid = blockIdx.x * 256 + threadIdx.x;
    int n = gid >> 6, d = gid & 63;
    int k = idx[n];
    float zq = emb[(size_t)k * CODE_DIM + d];
    float zz = z[gid];
    out_zq[gid] = zq;               // z_q_st == z_q value-wise
    float diff = zq - zz;
    atomicAdd(&dw[(size_t)k * CODE_DIM + d], zz);
    if (d == 0) atomicAdd(&counts[k], 1.0f);
    red[threadIdx.x] = diff * diff;
    __syncthreads();
    for (int s = 128; s > 0; s >>= 1) {
        if (threadIdx.x < s) red[threadIdx.x] += red[threadIdx.x + s];
        __syncthreads();
    }
    if (threadIdx.x == 0) atomicAdd(loss_accum, red[0]);
}

// new_cluster_size + global sum n
__global__ __launch_bounds__(256) void vq_csum_kernel(
    const float* __restrict__ ecs, const float* __restrict__ counts,
    float* __restrict__ out_cs, float* __restrict__ ncs_ws,
    float* __restrict__ nsum) {
    __shared__ float red[256];
    int k = blockIdx.x * 256 + threadIdx.x;
    float v = DECAY * ecs[k] + (1.0f - DECAY) * counts[k];
    out_cs[k] = v;
    ncs_ws[k] = v;
    red[threadIdx.x] = v;
    __syncthreads();
    for (int s = 128; s > 0; s >>= 1) {
        if (threadIdx.x < s) red[threadIdx.x] += red[threadIdx.x + s];
        __syncthreads();
    }
    if (threadIdx.x == 0) atomicAdd(nsum, red[0]);
}

// new_ema_embedding + normalized new_embedding
__global__ __launch_bounds__(256) void vq_final_kernel(
    const float* __restrict__ eee, const float* __restrict__ dw,
    const float* __restrict__ ncs, const float* __restrict__ nsum,
    float* __restrict__ out_eee, float* __restrict__ out_emb) {
    int i = blockIdx.x * 256 + threadIdx.x;
    int k = i >> 6;
    float nee = DECAY * eee[i] + (1.0f - DECAY) * dw[i];
    out_eee[i] = nee;
    float n = *nsum;
    float cs = (ncs[k] + EPSILON) / (n + (float)NUM_CODES * EPSILON) * n;
    out_emb[i] = nee / cs;
}

__global__ void vq_loss_kernel(const float* __restrict__ loss_accum,
                               float* __restrict__ out_loss) {
    if (threadIdx.x == 0)
        out_loss[0] = COMMIT * loss_accum[0] / (float)(N_ROWS * CODE_DIM);
}

// ---------------------------------------------------------------------------
extern "C" void kernel_launch(void* const* d_in, const int* in_sizes, int n_in,
                              void* d_out, int out_size, void* d_ws, size_t ws_size,
                              hipStream_t stream) {
    const float* z   = (const float*)d_in[0];
    const float* emb = (const float*)d_in[1];
    const float* ecs = (const float*)d_in[2];
    const float* eee = (const float*)d_in[3];
    float* out = (float*)d_out;
    float* ws  = (float*)d_ws;
    int*    idx_ws = (int*)(ws + WS_IDX);
    __bf16* ehi_ws = (__bf16*)((char*)d_ws + WS_EHI_BYTE);

    // 1. zero accumulators (loss, nsum, counts, ncs, dw)
    vq_zero_kernel<<<(WS_ZERON + 255) / 256, 256, 0, stream>>>(ws, WS_ZERON);

    // 2. bf16-round codebook + consistent squared norms
    vq_prep_kernel<<<NUM_CODES / 256, 256, 0, stream>>>(emb, ehi_ws, ws + WS_NORM);

    // 3. WMMA distance argmin (1024 waves x 32 rows; TDM-staged codebook)
    vq_argmin_kernel<<<N_ROWS / (32 * 8), 256, 0, stream>>>(
        z, ehi_ws, ws + WS_NORM, idx_ws, out + O_IDX);

    // 4. gather z_q, loss, scatter counts/dw
    vq_scatter_kernel<<<(N_ROWS * CODE_DIM) / 256, 256, 0, stream>>>(
        z, emb, idx_ws, out + O_ZQ, ws + WS_COUNTS, ws + WS_DW, ws + WS_LOSS);

    // 5. new_cluster_size + n
    vq_csum_kernel<<<NUM_CODES / 256, 256, 0, stream>>>(
        ecs, ws + WS_COUNTS, out + O_CS, ws + WS_NCS, ws + WS_NSUM);

    // 6. new_ema_embedding + new_embedding
    vq_final_kernel<<<(NUM_CODES * CODE_DIM) / 256, 256, 0, stream>>>(
        eee, ws + WS_DW, ws + WS_NCS, ws + WS_NSUM, out + O_EEE, out + O_EMB);

    // 7. loss scalar
    vq_loss_kernel<<<1, 32, 0, stream>>>(ws + WS_LOSS, out + O_LOSS);
}
